// AttentionHead_48077863911491
// MI455X (gfx1250) — compile-verified, matrix operand order checked by reading, not run on hardware
//
#include <hip/hip_runtime.h>
#include <hip/hip_bf16.h>

// ---------------------------------------------------------------------------
// AttentionHead for MI455X (gfx1250): bf16 WMMA everywhere, f32 accumulate.
//   B=4, S=2048, E=1024, D=64
//   q = h Wq + bq ; k = h Wk + bk   (v is dead code in the reference)
//   scores = (q k^T) / 8 ; weights = softmax over QUERY axis (columns)
//   out = weights @ h   <- dominant GEMM: async-to-LDS double-buffered WMMA
// ---------------------------------------------------------------------------

typedef __attribute__((ext_vector_type(16))) __bf16 v16bf;
typedef __attribute__((ext_vector_type(8)))  __bf16 v8bf;
typedef __attribute__((ext_vector_type(4)))  __bf16 v4bf;
typedef __attribute__((ext_vector_type(8)))  float  v8f;
typedef __attribute__((__vector_size__(16))) int    i32x4;

typedef __attribute__((address_space(1))) i32x4 gi32x4;   // global int4
typedef __attribute__((address_space(3))) i32x4 li32x4;   // LDS int4

#define BATCH 4
#define SEQ   2048
#define EMB   1024
#define HD    64
#define ROWS  (BATCH * SEQ)   // 8192

#define LDA_PAD 40            // LDS row stride in bf16 (80B: 16B-aligned, conflict-free)

// ---------------- WMMA helpers (wave32, 16x16x32 bf16 -> f32) ---------------

__device__ __forceinline__ v8f wmma_bf16(v16bf a, v16bf b, v8f c) {
  // (neg_a, A, neg_b, B, c_mod, C, reuse_a, reuse_b)
  return __builtin_amdgcn_wmma_f32_16x16x32_bf16(false, a, false, b,
                                                 (short)0, c, false, false);
}

// Load one 16x32 (MxK) bf16 WMMA operand from a row-major matrix with leading
// dimension ld (elements). Works for global or LDS pointers. Per the CDNA5
// 16-bit A layout: lane r (r=lane&15) holds row r; lanes 0-15 carry K 0..7 &
// 16..23, lanes 16-31 carry K 8..15 & 24..31 -> two 16B chunks per lane.
__device__ __forceinline__ v16bf load_op16x32(const __bf16* tile, int ld, int lane) {
  const int r  = lane & 15;
  const int kh = (lane >> 4) << 3;
  const __bf16* p = tile + (size_t)r * ld + kh;
  v8bf lo = *(const v8bf*)p;          // K = kh + 0..7   -> VGPR 0..3
  v8bf hi = *(const v8bf*)(p + 16);   // K = kh + 16..23 -> VGPR 4..7
  v16bf t;
#pragma unroll
  for (int i = 0; i < 8; ++i) { t[i] = lo[i]; t[i + 8] = hi[i]; }
  return t;
}

__device__ __forceinline__ void prefetch_op(const __bf16* tile, int ld, int lane) {
  const int r  = lane & 15;
  const int kh = (lane >> 4) << 3;
  __builtin_prefetch(tile + (size_t)r * ld + kh, 0, 1);  // global_prefetch_b8
}

// Async 16B global -> LDS copy (GLOBAL_LOAD_ASYNC_TO_LDS_B128, ASYNCcnt).
__device__ __forceinline__ void async_cp16(const __bf16* gsrc, __bf16* ldst) {
  __builtin_amdgcn_global_load_async_to_lds_b128(
      (gi32x4*)gsrc,
      (li32x4*)ldst,
      /*imm_offset=*/0, /*cpol=*/0);
}

// C/D layout: VGPR r holds row m0 + r + (lane<16 ? 0 : 8), column n0 + lane&15.
__device__ __forceinline__ void store_tile_f32(float* base, int ldc, int m0, int n0,
                                               v8f c, int lane, float scale) {
  const int col  = n0 + (lane & 15);
  const int row0 = m0 + ((lane >> 4) << 3);
#pragma unroll
  for (int r = 0; r < 8; ++r)
    base[(size_t)(row0 + r) * ldc + col] = c[r] * scale;
}

// ---------------- 1) f32 -> bf16 conversion of h ---------------------------

__global__ void k_cvt_h(const float4* __restrict__ h4, __bf16* __restrict__ hb) {
  size_t i = (size_t)blockIdx.x * blockDim.x + threadIdx.x;  // ROWS*EMB/4 threads
  float4 v = h4[i];
  v4bf o;
  o[0] = (__bf16)v.x; o[1] = (__bf16)v.y; o[2] = (__bf16)v.z; o[3] = (__bf16)v.w;
  *(v4bf*)(hb + i * 4) = o;
}

// ---------------- 2) transpose + convert Wq/Wk: [E,D] f32 -> [D,E] bf16 -----

__global__ void k_cvt_w(const float* __restrict__ Wq, const float* __restrict__ Wk,
                        __bf16* __restrict__ wqt, __bf16* __restrict__ wkt) {
  int i = blockIdx.x * blockDim.x + threadIdx.x;  // HD*EMB threads
  int d = i >> 10;          // / EMB
  int e = i & (EMB - 1);
  wqt[i] = (__bf16)Wq[(size_t)e * HD + d];
  wkt[i] = (__bf16)Wk[(size_t)e * HD + d];
}

// ---------------- 3) projection GEMM: [8192x1024]x[1024x64] + bias ----------
// Block = 256 threads = 8 waves arranged 2(M) x 4(N); block tile 32 x 64.

__global__ void k_proj(const __bf16* __restrict__ hb, const __bf16* __restrict__ wt,
                       const float* __restrict__ bias, __bf16* __restrict__ dst) {
  const int wave = threadIdx.x >> 5;
  const int lane = threadIdx.x & 31;
  const int m0 = blockIdx.x * 32 + (wave >> 2) * 16;
  const int n0 = (wave & 3) * 16;
  v8f acc = {};
#pragma unroll 2
  for (int k0 = 0; k0 < EMB; k0 += 32) {
    if (k0 + 128 < EMB)
      prefetch_op(hb + (size_t)m0 * EMB + k0 + 128, EMB, lane);
    v16bf a = load_op16x32(hb + (size_t)m0 * EMB + k0, EMB, lane);
    v16bf b = load_op16x32(wt + (size_t)n0 * EMB + k0, EMB, lane);
    acc = wmma_bf16(a, b, acc);
  }
  const int col  = n0 + (lane & 15);
  const int row0 = m0 + ((lane >> 4) << 3);
  const float bv = bias[col];
#pragma unroll
  for (int r = 0; r < 8; ++r)
    dst[(size_t)(row0 + r) * HD + col] = (__bf16)(acc[r] + bv);
}

// ---------------- 4) scores = q k^T * 0.125 : f32 [B][S][S] -----------------
// One 16x16 tile per wave; K loop over D=64 (2 WMMAs).

__global__ void k_scores(const __bf16* __restrict__ qb, const __bf16* __restrict__ kb,
                         float* __restrict__ scores) {
  const int lane = threadIdx.x & 31;
  const int gw = blockIdx.x * 8 + (threadIdx.x >> 5);
  const int b  = gw >> 14;          // / (128*128)
  const int t  = gw & 16383;
  const int mt = t >> 7, nt = t & 127;
  const __bf16* q = qb + (size_t)b * SEQ * HD;
  const __bf16* k = kb + (size_t)b * SEQ * HD;
  v8f acc = {};
#pragma unroll
  for (int k0 = 0; k0 < HD; k0 += 32) {
    v16bf a  = load_op16x32(q + (size_t)(mt * 16) * HD + k0, HD, lane);
    v16bf bb = load_op16x32(k + (size_t)(nt * 16) * HD + k0, HD, lane);
    acc = wmma_bf16(a, bb, acc);
  }
  store_tile_f32(scores + (size_t)b * SEQ * SEQ, SEQ, mt * 16, nt * 16,
                 acc, lane, 0.125f);
}

// ---------------- 5) column softmax (over query axis), write bf16 ----------

__global__ void k_softmax(const float* __restrict__ scores, __bf16* __restrict__ wts) {
  const int b   = blockIdx.y;
  const int col = blockIdx.x * blockDim.x + threadIdx.x;  // key index
  const float* s = scores + (size_t)b * SEQ * SEQ + col;
  float m = -3.4e38f;
  for (int q = 0; q < SEQ; ++q) m = fmaxf(m, s[(size_t)q * SEQ]);
  float sum = 0.f;
  for (int q = 0; q < SEQ; ++q) sum += __expf(s[(size_t)q * SEQ] - m);
  const float inv = 1.f / sum;
  __bf16* w = wts + (size_t)b * SEQ * SEQ + col;
  for (int q = 0; q < SEQ; ++q)
    w[(size_t)q * SEQ] = (__bf16)(__expf(s[(size_t)q * SEQ] - m) * inv);
}

// ---------------- 6) transpose h_bf: [B][S][E] -> [B][E][S] ----------------

__global__ void k_transpose(const __bf16* __restrict__ hb, __bf16* __restrict__ ht) {
  __shared__ __bf16 tile[32][33];
  const int b  = blockIdx.z;
  const int e0 = blockIdx.x * 32;
  const int s0 = blockIdx.y * 32;
  const int tx = threadIdx.x;
#pragma unroll
  for (int j = threadIdx.y; j < 32; j += 8)
    tile[j][tx] = hb[(size_t)(b * SEQ + s0 + j) * EMB + e0 + tx];
  __syncthreads();
#pragma unroll
  for (int j = threadIdx.y; j < 32; j += 8)
    ht[(size_t)(b * EMB + e0 + j) * SEQ + s0 + tx] = tile[tx][j];
}

// ---------------- 7) out = weights @ h : per batch [SxS]x[SxE] -------------
// Block tile 128(M) x 64(N), 8 waves (4M x 2N), each wave 32x32 (2x2 WMMA).
// K panels (A:128x32, B:64x32) stream into LDS with async-to-LDS copies,
// double-buffered, tracked by ASYNCcnt; waves consume via ds_load_b128.

__global__ void __launch_bounds__(256)
k_out(const __bf16* __restrict__ wts, const __bf16* __restrict__ ht,
      float* __restrict__ out) {
  __shared__ __align__(16) __bf16 lA[2][128][LDA_PAD];
  __shared__ __align__(16) __bf16 lB[2][64][LDA_PAD];

  const int b    = blockIdx.z;
  const int t    = threadIdx.x;
  const int wave = t >> 5;
  const int lane = t & 31;
  const int m0b  = blockIdx.x * 128;
  const int n0b  = blockIdx.y * 64;
  const int mL   = (wave >> 1) * 32;   // wave tile origin inside block tile
  const int nL   = (wave & 1) * 32;

  const __bf16* A  = wts + (size_t)b * SEQ * SEQ;   // [S(q) x S(k)] row-major
  const __bf16* Bt = ht  + (size_t)b * EMB * SEQ;   // [E x S(k)]   row-major

  // Per-thread 16B chunk coords: 4 chunks per 32-wide row.
  const int row = t >> 2;          // 0..63
  const int col = (t & 3) * 8;     // element offset 0,8,16,24

  // Issue the 3 async copies (2 for A's 128 rows, 1 for B's 64 rows).
  auto load_panels = [&](int buf, int k0) {
    async_cp16(A  + (size_t)(m0b + row)      * SEQ + k0 + col, &lA[buf][row][col]);
    async_cp16(A  + (size_t)(m0b + 64 + row) * SEQ + k0 + col, &lA[buf][64 + row][col]);
    async_cp16(Bt + (size_t)(n0b + row)      * SEQ + k0 + col, &lB[buf][row][col]);
  };

  v8f c00 = {}, c01 = {}, c10 = {}, c11 = {};
  load_panels(0, 0);
  int buf = 0;
  for (int k0 = 0; k0 < SEQ; k0 += 32) {
    if (k0 + 32 < SEQ) {
      load_panels(buf ^ 1, k0 + 32);              // prefetch next panel
      __builtin_amdgcn_s_wait_asynccnt(3);        // wait for current panel only
    } else {
      __builtin_amdgcn_s_wait_asynccnt(0);
    }
    __syncthreads();                              // panel visible to all waves
    v16bf a0 = load_op16x32(&lA[buf][mL][0],      LDA_PAD, lane);
    v16bf a1 = load_op16x32(&lA[buf][mL + 16][0], LDA_PAD, lane);
    v16bf b0 = load_op16x32(&lB[buf][nL][0],      LDA_PAD, lane);
    v16bf b1 = load_op16x32(&lB[buf][nL + 16][0], LDA_PAD, lane);
    c00 = wmma_bf16(a0, b0, c00);
    c01 = wmma_bf16(a0, b1, c01);
    c10 = wmma_bf16(a1, b0, c10);
    c11 = wmma_bf16(a1, b1, c11);
    __syncthreads();                              // done reading before overwrite
    buf ^= 1;
  }

  float* ob = out + (size_t)b * SEQ * EMB;
  const int m0 = m0b + mL, n0 = n0b + nL;
  store_tile_f32(ob, EMB, m0,      n0,      c00, lane, 1.f);
  store_tile_f32(ob, EMB, m0,      n0 + 16, c01, lane, 1.f);
  store_tile_f32(ob, EMB, m0 + 16, n0,      c10, lane, 1.f);
  store_tile_f32(ob, EMB, m0 + 16, n0 + 16, c11, lane, 1.f);
}

// ---------------- launch ---------------------------------------------------

extern "C" void kernel_launch(void* const* d_in, const int* in_sizes, int n_in,
                              void* d_out, int out_size, void* d_ws, size_t ws_size,
                              hipStream_t stream) {
  const float* h  = (const float*)d_in[0];
  const float* Wq = (const float*)d_in[1];
  const float* bq = (const float*)d_in[2];
  const float* Wk = (const float*)d_in[3];
  const float* bk = (const float*)d_in[4];
  // d_in[5] (Wv) and d_in[6] (bv) are dead in the reference.
  float* out = (float*)d_out;

  // Workspace layout (all offsets 256B-aligned)
  char* wsb = (char*)d_ws;
  constexpr size_t SZ_HBF = (size_t)ROWS * EMB * 2;        // 16 MiB
  constexpr size_t SZ_HT  = SZ_HBF;                        // 16 MiB
  constexpr size_t SZ_QK  = (size_t)ROWS * HD * 2;         // 1 MiB each
  constexpr size_t SZ_WT  = (size_t)HD * EMB * 2;          // 128 KiB each
  constexpr size_t SZ_SC  = (size_t)BATCH * SEQ * SEQ * 4; // 64 MiB
  size_t off = 0;
  __bf16* h_bf = (__bf16*)(wsb + off); off += SZ_HBF;
  __bf16* ht   = (__bf16*)(wsb + off); off += SZ_HT;
  __bf16* q_bf = (__bf16*)(wsb + off); off += SZ_QK;
  __bf16* k_bf = (__bf16*)(wsb + off); off += SZ_QK;
  __bf16* wqt  = (__bf16*)(wsb + off); off += SZ_WT;
  __bf16* wkt  = (__bf16*)(wsb + off); off += SZ_WT;
  float*  sc   = (float*) (wsb + off); off += SZ_SC;
  __bf16* wts  = (__bf16*)(wsb + off);

  // 1) h -> bf16
  k_cvt_h<<<(ROWS * EMB / 4) / 256, 256, 0, stream>>>((const float4*)h, h_bf);
  // 2) Wq/Wk -> transposed bf16
  k_cvt_w<<<(HD * EMB) / 256, 256, 0, stream>>>(Wq, Wk, wqt, wkt);
  // 3) projections (WMMA)
  k_proj<<<ROWS / 32, 256, 0, stream>>>(h_bf, wqt, bq, q_bf);
  k_proj<<<ROWS / 32, 256, 0, stream>>>(h_bf, wkt, bk, k_bf);
  // 4) scores (WMMA)
  k_scores<<<(BATCH * 128 * 128) / 8, 256, 0, stream>>>(q_bf, k_bf, sc);
  // 5) column softmax
  k_softmax<<<dim3(SEQ / 256, BATCH), 256, 0, stream>>>(sc, wts);
  // 6) h^T for the B operand of the output GEMM
  k_transpose<<<dim3(EMB / 32, SEQ / 32, BATCH), dim3(32, 8), 0, stream>>>(h_bf, ht);
  // 7) out = weights @ h (WMMA, async-to-LDS double buffered)
  k_out<<<dim3(SEQ / 128, EMB / 64, BATCH), 256, 0, stream>>>(wts, ht, out);
}
